// MyKNIFE_46420006535557
// MI455X (gfx1250) — matrix operand order repaired and checked by reading, not compiled, not run on hardware
//
#include <hip/hip_runtime.h>
#include <hip/hip_bf16.h>
#include <cstdint>
#include <cstddef>

typedef _Float16 v16h __attribute__((ext_vector_type(16)));
typedef float    v8f  __attribute__((ext_vector_type(8)));

#define B_TOTAL   131072
#define WAVES     4
#define BLOCKS_N  (B_TOTAL / (WAVES * 16))   // 2048
#define LOG2PI_F  1.8378770664093453f

// ---- packed-weight tile bases (each tile = 32 lanes * 16 halves = 512 halves) ----
enum {
  T_MC = 0, T_MM0 = 2, T_MM1 = 4, T_MM2 = 6,
  T_ML0 = 14, T_ML1 = 16, T_ML2 = 18,
  T_MW0 = 26, T_MW1 = 28,
  T_CC = 29,
  T_CM0 = 31, T_CM1 = 39, T_CM2 = 47,
  T_CL0 = 79, T_CL1 = 87, T_CL2 = 95,
  T_CW0 = 127, T_CW1 = 135, T_CW2 = 143,
  T_TOTAL = 147
};

struct Params {
  const float *x, *phi; const int *y;
  const float *mcW, *mcB;
  const float *mm0W, *mm0B, *mm1W, *mm1B, *mm2W, *mm2B;
  const float *ml0W, *ml0B, *ml1W, *ml1B, *ml2W, *ml2B;
  const float *mw0W, *mw0B, *mw1W, *mw1B;
  const float *ccW, *ccB;
  const float *cm0W, *cm0B, *cm1W, *cm1B, *cm2W, *cm2B;
  const float *cl0W, *cl0B, *cl1W, *cl1B, *cl2W, *cl2B;
  const float *cw0W, *cw0B, *cw1W, *cw1B, *cw2W, *cw2B;
  _Float16* pk;       // packed B-operand weights (f16)
  float* partials;    // [BLOCKS_N][2]
};

// Wave-local LDS fence: each wave's slab is private, and LDS ops from one wave
// complete in order (DScnt). s_wait_dscnt 0 + memory clobber stops the compiler
// from caching/reordering LDS accesses across the cross-lane handoff, without
// convoying the other waves of the WGP the way s_barrier would.
__device__ __forceinline__ void waveFence() {
  asm volatile("s_wait_dscnt 0" ::: "memory");
}

// Fast tanh via hardware exp: tanh(x) = 1 - 2/(e^{2x}+1); exact limits at +/-inf.
__device__ __forceinline__ float fastTanh(float x) {
  return 1.0f - 2.0f / (__expf(2.0f * x) + 1.0f);
}

// -------------------- weight pre-pack (f16, B-operand lane layout) --------------------
// B 32x16 f16 layout: lanes 0-15 hold N=lane with K={0..7,16..23} (2 K per VGPR),
// lanes 16-31 hold N=lane-16 with K={8..15,24..31}.
__global__ void knife_pack(Params p) {
  const int g = blockIdx.x, l = threadIdx.x;
  const float* W = nullptr; int Nout = 0, base = 0, heads = 1;
  switch (g) {
    case 0:  W = p.mcW;  Nout = 32;  base = T_MC;  break;
    case 1:  W = p.mm0W; Nout = 32;  base = T_MM0; break;
    case 2:  W = p.mm1W; Nout = 32;  base = T_MM1; break;
    case 3:  W = p.mm2W; Nout = 128; base = T_MM2; break;
    case 4:  W = p.ml0W; Nout = 32;  base = T_ML0; break;
    case 5:  W = p.ml1W; Nout = 32;  base = T_ML1; break;
    case 6:  W = p.ml2W; Nout = 128; base = T_ML2; break;
    case 7:  W = p.mw0W; Nout = 32;  base = T_MW0; break;
    case 8:  W = p.mw1W; Nout = 4;   base = T_MW1; break;
    case 9:  W = p.ccW;  Nout = 32;  base = T_CC;  break;
    case 10: W = p.cm0W; Nout = 32;  base = T_CM0; heads = 4; break;
    case 11: W = p.cm1W; Nout = 32;  base = T_CM1; heads = 4; break;
    case 12: W = p.cm2W; Nout = 128; base = T_CM2; heads = 4; break;
    case 13: W = p.cl0W; Nout = 32;  base = T_CL0; heads = 4; break;
    case 14: W = p.cl1W; Nout = 32;  base = T_CL1; heads = 4; break;
    case 15: W = p.cl2W; Nout = 128; base = T_CL2; heads = 4; break;
    case 16: W = p.cw0W; Nout = 32;  base = T_CW0; heads = 4; break;
    case 17: W = p.cw1W; Nout = 32;  base = T_CW1; heads = 4; break;
    default: W = p.cw2W; Nout = 4;   base = T_CW2; heads = 4; break;
  }
  const int tilesPerHead = (Nout + 15) >> 4;
  const int n = l & 15;
  const int kofs = (l & 16) ? 8 : 0;
  for (int h = 0; h < heads; ++h) {
    const float* Wh = W + (size_t)h * 32 * Nout;
    for (int t = 0; t < tilesPerHead; ++t) {
      const int ng = t * 16 + n;
      _Float16* dst = p.pk + ((size_t)(base + h * tilesPerHead + t) * 32 + l) * 16;
      #pragma unroll
      for (int j = 0; j < 16; ++j) {
        const int K = (j < 8 ? j : j + 8) + kofs;
        const float v = (ng < Nout) ? Wh[K * Nout + ng] : 0.0f;
        dst[j] = (_Float16)v;
      }
    }
  }
}

// -------------------- per-wave LDS slab --------------------
struct alignas(16) Slab {
  float    phi[16][32];     // staged phi (f32)
  float    wbuf[16][4];     // mixture logits (f32)
  _Float16 zc[16][32];      // conv output
  _Float16 t0[16][32];      // ping
  _Float16 t1[16][32];      // pong
  _Float16 mu[16][128];
  _Float16 lv[16][128];     // tanh already applied
  int      yv[16];
};

__device__ __forceinline__ uint32_t relu2(uint32_t u) {
  uint32_t lo = u & 0xFFFFu;  if (lo & 0x8000u) lo = 0;
  uint32_t hi = u >> 16;      if (hi & 0x8000u) hi = 0;
  return lo | (hi << 16);
}

// A operand (16-bit A 16x32): lane<16 -> M=lane, K in {0..7,16..23};
// lane>=16 -> M=lane-16, K in {8..15,24..31}; 2 consecutive K per dword.
__device__ __forceinline__ v16h buildA(const _Float16* in, int lane) {
  union { v16h v; uint32_t u[8]; } A;
  const int m = lane & 15;
  const int hi4 = (lane & 16) ? 4 : 0;
  const uint32_t* rp = (const uint32_t*)(in + m * 32);
  #pragma unroll
  for (int t = 0; t < 8; ++t) {
    const int dw = t + (t >= 4 ? 4 : 0) + hi4;
    A.u[t] = relu2(rp[dw]);       // fused ReLU (layer inputs are always ReLU'd)
  }
  return A.v;
}

// One linear layer for 16 rows: D = A(16x32) x W(32x16*ntiles) + bias, store per C/D layout.
// mode: 0 = f16 store, 1 = f16 tanh store, 2 = f32 store. head>=0: store only rows with y==head.
__device__ __forceinline__ void doLayer(v16h a, const _Float16* pk, int tile0, int ntiles,
                                        const float* bias, int Nout,
                                        _Float16* out16, float* out32, int stride, int mode,
                                        const int* y8, int head, int lane) {
  const int n = lane & 15;
  const int mbase = (lane & 16) ? 8 : 0;
  for (int t = 0; t < ntiles; ++t) {
    const _Float16* bp = pk + ((size_t)(tile0 + t) * 32 + lane) * 16;
    union { v16h v; uint4 q[2]; } Bv;
    Bv.q[0] = *(const uint4*)bp;
    Bv.q[1] = *(const uint4*)(bp + 8);
    union { v8f v; float f[8]; } C;
    #pragma unroll
    for (int i = 0; i < 8; ++i) C.f[i] = 0.0f;
    C.v = __builtin_amdgcn_wmma_f32_16x16x32_f16(false, a, false, Bv.v, (short)0, C.v, false, false);
    const int ng = t * 16 + n;
    if (ng < Nout) {
      const float beta = bias[ng];
      #pragma unroll
      for (int r = 0; r < 8; ++r) {
        if (head >= 0 && y8[r] != head) continue;
        const float v = C.f[r] + beta;
        const int M = mbase + r;
        if (mode == 2)      out32[M * stride + ng] = v;
        else if (mode == 1) out16[M * stride + ng] = (_Float16)fastTanh(v);
        else                out16[M * stride + ng] = (_Float16)v;
      }
    }
  }
}

// Gaussian-mixture log-pdf; lanes l and l+16 share row m=l&15 (2 components each).
// mu/lv rows read as b128 chunks, phi as float4 (all 16B aligned).
// logvar is tanh-bounded, so log(exp(lv)+1e-8) == lv to ~3e-8 abs: drop the log.
__device__ __forceinline__ float gmAccum(const Slab& S, int lane) {
  const int m = lane & 15;
  const int kb = (lane & 16) ? 2 : 0;
  const float w0 = S.wbuf[m][0], w1 = S.wbuf[m][1], w2 = S.wbuf[m][2], w3 = S.wbuf[m][3];
  const float mx = fmaxf(fmaxf(w0, w1), fmaxf(w2, w3));
  const float lse_w = mx + __logf(__expf(w0 - mx) + __expf(w1 - mx) + __expf(w2 - mx) + __expf(w3 - mx));
  float s0 = 0.f, s1 = 0.f;
  #pragma unroll
  for (int kk = 0; kk < 2; ++kk) {
    const int k = kb + kk;
    float ss = 0.f, sl = 0.f;
    #pragma unroll
    for (int c = 0; c < 4; ++c) {
      union { uint4 q; _Float16 h[8]; } MU, LV;
      MU.q = *(const uint4*)&S.mu[m][k * 32 + c * 8];
      LV.q = *(const uint4*)&S.lv[m][k * 32 + c * 8];
      union { float4 v; float f[4]; } P0, P1;
      P0.v = *(const float4*)&S.phi[m][c * 8];
      P1.v = *(const float4*)&S.phi[m][c * 8 + 4];
      #pragma unroll
      for (int j = 0; j < 8; ++j) {
        const float muv = (float)MU.h[j];
        const float lvv = (float)LV.h[j];
        const float var = __expf(lvv);          // v_exp_f32 (TRANS, co-executes)
        const float ph  = (j < 4) ? P0.f[j] : P1.f[j - 4];
        const float z = (ph - muv) * var;
        ss += z * z;
        sl += lvv;                              // == log(var + 1e-8) within 3e-8
      }
    }
    const float wk = (k == 0 ? w0 : k == 1 ? w1 : k == 2 ? w2 : w3) - lse_w;
    const float sv = -0.5f * ss + sl + wk;
    if (kk == 0) s0 = sv; else s1 = sv;
  }
  const float M1 = fmaxf(s0, s1);
  const float e1 = __expf(s0 - M1) + __expf(s1 - M1);
  const float M2 = __shfl_xor(M1, 16, 32);
  const float e2 = __shfl_xor(e1, 16, 32);
  const float Mg = fmaxf(M1, M2);
  const float tot = e1 * __expf(M1 - Mg) + e2 * __expf(M2 - Mg);
  const float lp = (-0.5f * 32.0f * LOG2PI_F) + Mg + __logf(tot);
  return (lane < 16) ? -lp : 0.0f;
}

// -------------------- main kernel --------------------
__global__ void __launch_bounds__(WAVES * 32) knife_main(Params p) {
  __shared__ Slab slabs[WAVES];
  __shared__ float blkRed[WAVES * 2];
  const int tid = threadIdx.x;
  const int wave = tid >> 5;
  const int lane = tid & 31;
  Slab& S = slabs[wave];
  const long rowbase = (long)(blockIdx.x * WAVES + wave) * 16;

  { // stage phi (f32) and y
    const float* src = p.phi + rowbase * 32;
    float* dst = &S.phi[0][0];
    #pragma unroll
    for (int i = 0; i < 16; ++i) dst[lane + i * 32] = src[lane + i * 32];
    if (lane < 16) S.yv[lane] = p.y[rowbase + lane];
  }

  // A operand for x (no ReLU), built straight from global per the A layout.
  v16h A0;
  {
    union { v16h v; _Float16 h[16]; } A;
    const int m = lane & 15;
    const int hi = (lane & 16) ? 8 : 0;
    const float* xr = p.x + (rowbase + m) * 32;
    #pragma unroll
    for (int t = 0; t < 8; ++t) {
      const int k = 2 * t + (t >= 4 ? 8 : 0) + hi;
      A.h[2 * t]     = (_Float16)xr[k];
      A.h[2 * t + 1] = (_Float16)xr[k + 1];
    }
    A0 = A.v;
  }
  waveFence();

  // ================= marginal kernel =================
  doLayer(A0, p.pk, T_MC, 2, p.mcB, 32, &S.zc[0][0], nullptr, 32, 0, nullptr, -1, lane);
  waveFence();
  const v16h Az = buildA(&S.zc[0][0], lane);
  // mu
  doLayer(Az, p.pk, T_MM0, 2, p.mm0B, 32, &S.t0[0][0], nullptr, 32, 0, nullptr, -1, lane);
  waveFence();
  v16h Ah = buildA(&S.t0[0][0], lane);
  doLayer(Ah, p.pk, T_MM1, 2, p.mm1B, 32, &S.t1[0][0], nullptr, 32, 0, nullptr, -1, lane);
  waveFence();
  Ah = buildA(&S.t1[0][0], lane);
  doLayer(Ah, p.pk, T_MM2, 8, p.mm2B, 128, &S.mu[0][0], nullptr, 128, 0, nullptr, -1, lane);
  waveFence();
  // logvar (tanh on output)
  doLayer(Az, p.pk, T_ML0, 2, p.ml0B, 32, &S.t0[0][0], nullptr, 32, 0, nullptr, -1, lane);
  waveFence();
  Ah = buildA(&S.t0[0][0], lane);
  doLayer(Ah, p.pk, T_ML1, 2, p.ml1B, 32, &S.t1[0][0], nullptr, 32, 0, nullptr, -1, lane);
  waveFence();
  Ah = buildA(&S.t1[0][0], lane);
  doLayer(Ah, p.pk, T_ML2, 8, p.ml2B, 128, &S.lv[0][0], nullptr, 128, 1, nullptr, -1, lane);
  waveFence();
  // mixture weights
  doLayer(Az, p.pk, T_MW0, 2, p.mw0B, 32, &S.t0[0][0], nullptr, 32, 0, nullptr, -1, lane);
  waveFence();
  Ah = buildA(&S.t0[0][0], lane);
  doLayer(Ah, p.pk, T_MW1, 1, p.mw1B, 4, nullptr, &S.wbuf[0][0], 4, 2, nullptr, -1, lane);
  waveFence();

  const float margAcc = gmAccum(S, lane);
  waveFence();

  // ================= conditional kernel (label-routed heads) =================
  doLayer(A0, p.pk, T_CC, 2, p.ccB, 32, &S.zc[0][0], nullptr, 32, 0, nullptr, -1, lane);
  waveFence();
  const v16h Az2 = buildA(&S.zc[0][0], lane);
  int y8[8];
  {
    const int mb = (lane & 16) ? 8 : 0;
    #pragma unroll
    for (int r = 0; r < 8; ++r) y8[r] = S.yv[mb + r];
  }
  #pragma unroll 1
  for (int h = 0; h < 4; ++h) {
    // mu head
    doLayer(Az2, p.pk, T_CM0 + 2 * h, 2, p.cm0B + h * 32, 32, &S.t0[0][0], nullptr, 32, 0, nullptr, -1, lane);
    waveFence();
    Ah = buildA(&S.t0[0][0], lane);
    doLayer(Ah, p.pk, T_CM1 + 2 * h, 2, p.cm1B + h * 32, 32, &S.t1[0][0], nullptr, 32, 0, nullptr, -1, lane);
    waveFence();
    Ah = buildA(&S.t1[0][0], lane);
    doLayer(Ah, p.pk, T_CM2 + 8 * h, 8, p.cm2B + h * 128, 128, &S.mu[0][0], nullptr, 128, 0, y8, h, lane);
    waveFence();
    // logvar head (tanh)
    doLayer(Az2, p.pk, T_CL0 + 2 * h, 2, p.cl0B + h * 32, 32, &S.t0[0][0], nullptr, 32, 0, nullptr, -1, lane);
    waveFence();
    Ah = buildA(&S.t0[0][0], lane);
    doLayer(Ah, p.pk, T_CL1 + 2 * h, 2, p.cl1B + h * 32, 32, &S.t1[0][0], nullptr, 32, 0, nullptr, -1, lane);
    waveFence();
    Ah = buildA(&S.t1[0][0], lane);
    doLayer(Ah, p.pk, T_CL2 + 8 * h, 8, p.cl2B + h * 128, 128, &S.lv[0][0], nullptr, 128, 1, y8, h, lane);
    waveFence();
    // weight head (3 layers)
    doLayer(Az2, p.pk, T_CW0 + 2 * h, 2, p.cw0B + h * 32, 32, &S.t0[0][0], nullptr, 32, 0, nullptr, -1, lane);
    waveFence();
    Ah = buildA(&S.t0[0][0], lane);
    doLayer(Ah, p.pk, T_CW1 + 2 * h, 2, p.cw1B + h * 32, 32, &S.t1[0][0], nullptr, 32, 0, nullptr, -1, lane);
    waveFence();
    Ah = buildA(&S.t1[0][0], lane);
    doLayer(Ah, p.pk, T_CW2 + h, 1, p.cw2B + h * 4, 4, nullptr, &S.wbuf[0][0], 4, 2, y8, h, lane);
    waveFence();
  }
  const float condAcc = gmAccum(S, lane);

  // ---- deterministic block reduction (only place a real barrier is needed) ----
  float mv = margAcc, cv = condAcc;
  #pragma unroll
  for (int off = 16; off > 0; off >>= 1) {
    mv += __shfl_xor(mv, off, 32);
    cv += __shfl_xor(cv, off, 32);
  }
  if (lane == 0) { blkRed[wave * 2] = mv; blkRed[wave * 2 + 1] = cv; }
  __syncthreads();
  if (tid == 0) {
    float sm = 0.f, sc = 0.f;
    for (int w = 0; w < WAVES; ++w) { sm += blkRed[w * 2]; sc += blkRed[w * 2 + 1]; }
    p.partials[blockIdx.x * 2]     = sm;
    p.partials[blockIdx.x * 2 + 1] = sc;
  }
}

// -------------------- finalize --------------------
__global__ void knife_finalize(const float* partials, float* out) {
  __shared__ float sm[256], sc[256];
  float a = 0.f, b = 0.f;
  for (int i = threadIdx.x; i < BLOCKS_N; i += 256) {
    a += partials[2 * i];
    b += partials[2 * i + 1];
  }
  sm[threadIdx.x] = a; sc[threadIdx.x] = b;
  __syncthreads();
  for (int s = 128; s > 0; s >>= 1) {
    if ((int)threadIdx.x < s) {
      sm[threadIdx.x] += sm[threadIdx.x + s];
      sc[threadIdx.x] += sc[threadIdx.x + s];
    }
    __syncthreads();
  }
  if (threadIdx.x == 0) {
    const float marg = sm[0] / (float)B_TOTAL;
    const float cond = sc[0] / (float)B_TOTAL;
    out[0] = marg - cond;
    out[1] = marg;
    out[2] = cond;
  }
}

// -------------------- host launch --------------------
extern "C" void kernel_launch(void* const* d_in, const int* in_sizes, int n_in,
                              void* d_out, int out_size, void* d_ws, size_t ws_size,
                              hipStream_t stream) {
  auto F = [&](int i) { return (const float*)d_in[i]; };
  Params p{};
  const bool jaxOrder = (in_sizes[0] == 1024);  // pytree: sorted keys -> cond.conv.W first
  if (jaxOrder) {
    p.ccW  = F(0);  p.ccB  = F(1);
    p.cl0W = F(2);  p.cl0B = F(3);  p.cl1W = F(4);  p.cl1B = F(5);  p.cl2W = F(6);  p.cl2B = F(7);
    p.cm0W = F(8);  p.cm0B = F(9);  p.cm1W = F(10); p.cm1B = F(11); p.cm2W = F(12); p.cm2B = F(13);
    p.cw0W = F(14); p.cw0B = F(15); p.cw1W = F(16); p.cw1B = F(17); p.cw2W = F(18); p.cw2B = F(19);
    p.mcW  = F(20); p.mcB  = F(21);
    p.ml0W = F(22); p.ml0B = F(23); p.ml1W = F(24); p.ml1B = F(25); p.ml2W = F(26); p.ml2B = F(27);
    p.mm0W = F(28); p.mm0B = F(29); p.mm1W = F(30); p.mm1B = F(31); p.mm2W = F(32); p.mm2B = F(33);
    p.mw0W = F(34); p.mw0B = F(35); p.mw1W = F(36); p.mw1B = F(37);
    p.phi = F(38); p.x = F(39); p.y = (const int*)d_in[40];
  } else {                                       // insertion order: x, phi, y, params...
    p.x = F(0); p.phi = F(1); p.y = (const int*)d_in[2];
    p.mcW  = F(3);  p.mcB  = F(4);
    p.mm0W = F(5);  p.mm0B = F(6);  p.mm1W = F(7);  p.mm1B = F(8);  p.mm2W = F(9);  p.mm2B = F(10);
    p.ml0W = F(11); p.ml0B = F(12); p.ml1W = F(13); p.ml1B = F(14); p.ml2W = F(15); p.ml2B = F(16);
    p.mw0W = F(17); p.mw0B = F(18); p.mw1W = F(19); p.mw1B = F(20);
    p.ccW  = F(21); p.ccB  = F(22);
    p.cm0W = F(23); p.cm0B = F(24); p.cm1W = F(25); p.cm1B = F(26); p.cm2W = F(27); p.cm2B = F(28);
    p.cl0W = F(29); p.cl0B = F(30); p.cl1W = F(31); p.cl1B = F(32); p.cl2W = F(33); p.cl2B = F(34);
    p.cw0W = F(35); p.cw0B = F(36); p.cw1W = F(37); p.cw1B = F(38); p.cw2W = F(39); p.cw2B = F(40);
  }
  p.partials = (float*)d_ws;                                // 2048*2 f32 = 16 KB
  p.pk = (_Float16*)((char*)d_ws + 16384);                  // 147 tiles * 512 halves = 147 KB

  knife_pack<<<19, 32, 0, stream>>>(p);
  knife_main<<<BLOCKS_N, WAVES * 32, 0, stream>>>(p);
  knife_finalize<<<1, 256, 0, stream>>>(p.partials, (float*)d_out);
}
